// GDN_1d_7069516169777
// MI455X (gfx1250) — matrix operand order repaired and checked by reading, not compiled, not run on hardware
//
#include <hip/hip_runtime.h>
#include <hip/hip_bf16.h>

typedef __attribute__((ext_vector_type(16))) __bf16 v16bf;
typedef __attribute__((ext_vector_type(8)))  __bf16 v8bf;
typedef __attribute__((ext_vector_type(8)))  float  v8f;

#define C_DIM 256
#define L_DIM 16384
#define B_DIM 16
#define TN 64          // columns (l positions) per block
#define LDA 264        // padded LDS row stride in bf16 elements (528 B = 33*16B)

#define PEDESTAL_F     1.4551915228366852e-11f   // (2^-18)^2 = 2^-36
#define GAMMA_BOUND_F  3.0517578125e-05f         // 2^-15
#define BETA_BOUND_F   1.0000072759132946e-03f   // sqrt(1e-6 + pedestal)

__device__ __forceinline__ unsigned short f32_to_bf16(float f) {
    unsigned int u = __builtin_bit_cast(unsigned int, f);
    u += 0x7FFFu + ((u >> 16) & 1u);              // round-to-nearest-even
    return (unsigned short)(u >> 16);
}

// ---------------- prep: reparametrize gamma -> bf16, beta -> f32 ----------------
__global__ void gdn_prep_kernel(const float* __restrict__ beta_p,
                                const float* __restrict__ gamma_p,
                                unsigned short* __restrict__ gamma_bf,
                                float* __restrict__ beta_f) {
    int idx = blockIdx.x * blockDim.x + threadIdx.x;
    if (idx < C_DIM * C_DIM) {
        float p = gamma_p[idx];
        float q = fmaxf(p, GAMMA_BOUND_F);
        gamma_bf[idx] = f32_to_bf16(q * q - PEDESTAL_F);
    }
    if (idx < C_DIM) {
        float p = beta_p[idx];
        float q = fmaxf(p, BETA_BOUND_F);
        beta_f[idx] = q * q - PEDESTAL_F;
    }
}

// ---------------- main: norm = gamma @ x^2 + beta ; out = x * rsqrt(norm) ------
__global__ void __launch_bounds__(256)
gdn_main_kernel(const float* __restrict__ x,
                const unsigned short* __restrict__ gamma_bf,
                const float* __restrict__ beta_f,
                float* __restrict__ out) {
    extern __shared__ char smem[];
    unsigned short* gamma_s = (unsigned short*)smem;                       // [256][LDA]
    unsigned short* s_s = (unsigned short*)(smem + C_DIM * LDA * 2);       // [TN][LDA]

    const int tid    = threadIdx.x;
    const int lane   = tid & 31;
    const int wave   = tid >> 5;
    const int laneLo = lane & 15;
    const int laneHi = lane >> 4;

    const int b  = blockIdx.x >> 8;          // / (L/TN)
    const int l0 = (blockIdx.x & 255) * TN;

    // ---- stage gamma (bf16) into LDS with padded rows (coalesced b128 copies) ----
    for (int i = tid; i < (C_DIM * C_DIM) / 8; i += 256) {
        int e   = i * 8;
        int row = e >> 8;
        int col = e & 255;
        uint4 v = *reinterpret_cast<const uint4*>(gamma_bf + e);
        *reinterpret_cast<uint4*>(gamma_s + row * LDA + col) = v;
    }

    // ---- stage s = bf16(x^2), transposed to [col][chan] for B-fragment loads ----
    {
        const int c = tid;                   // one channel per thread
        const float* xr = x + ((size_t)b * C_DIM + c) * L_DIM + l0;
        #pragma unroll
        for (int v = 0; v < TN / 4; ++v) {
            float4 f = reinterpret_cast<const float4*>(xr)[v];
            int col = v * 4;
            s_s[(col + 0) * LDA + c] = f32_to_bf16(f.x * f.x);
            s_s[(col + 1) * LDA + c] = f32_to_bf16(f.y * f.y);
            s_s[(col + 2) * LDA + c] = f32_to_bf16(f.z * f.z);
            s_s[(col + 3) * LDA + c] = f32_to_bf16(f.w * f.w);
        }
    }
    __syncthreads();

    // ---- wave tiling: 8 waves = 4 (M) x 2 (N); each wave 64x32 output ----
    const int Mbase = (wave & 3) * 64;
    const int Nbase = (wave >> 2) * 32;

    v8f acc[4][2] = {};

    #pragma unroll
    for (int kt = 0; kt < 8; ++kt) {
        const int kb = kt * 32;

        // A fragments (gamma): 16-bit A 16x32 layout:
        //   lanes 0-15 hold K=[0..7] and K=[16..23]; lanes 16-31 hold K=[8..15],[24..31]
        v16bf afrag[4];
        #pragma unroll
        for (int i = 0; i < 4; ++i) {
            const v8bf* pa = reinterpret_cast<const v8bf*>(
                gamma_s + (Mbase + 16 * i + laneLo) * LDA + kb + laneHi * 8);
            v8bf lo = pa[0];     // K = k0 .. k0+7
            v8bf hi = pa[2];     // K = k0+16 .. k0+23
            afrag[i] = __builtin_shufflevector(lo, hi,
                0, 1, 2, 3, 4, 5, 6, 7, 8, 9, 10, 11, 12, 13, 14, 15);
        }

        // B fragments (x^2): 16-bit B 32x16 layout:
        //   lanes 0-15 -> col N=lane, K=[0..7],[16..23] ; lanes 16-31 -> same cols, K=[8..15],[24..31]
        v16bf bfrag[2];
        #pragma unroll
        for (int j = 0; j < 2; ++j) {
            const v8bf* pb = reinterpret_cast<const v8bf*>(
                s_s + (Nbase + 16 * j + laneLo) * LDA + kb + laneHi * 8);
            bfrag[j] = __builtin_shufflevector(pb[0], pb[2],
                0, 1, 2, 3, 4, 5, 6, 7, 8, 9, 10, 11, 12, 13, 14, 15);
        }

        #pragma unroll
        for (int i = 0; i < 4; ++i)
            #pragma unroll
            for (int j = 0; j < 2; ++j)
                acc[i][j] = __builtin_amdgcn_wmma_f32_16x16x32_bf16(
                    false, afrag[i], false, bfrag[j],
                    (short)0, acc[i][j], false, false);
    }

    // ---- epilogue: out = x * rsqrt(acc + beta)  (C/D: M = r + 8*laneHi, N = laneLo)
    #pragma unroll
    for (int i = 0; i < 4; ++i) {
        const int mb = Mbase + 16 * i;
        #pragma unroll
        for (int j = 0; j < 2; ++j) {
            const int lcol = Nbase + 16 * j + laneLo;
            const size_t colAddr = (size_t)b * C_DIM * L_DIM + (size_t)(l0 + lcol);
            #pragma unroll
            for (int r = 0; r < 8; ++r) {
                const int m = mb + r + 8 * laneHi;
                const size_t idx = colAddr + (size_t)m * L_DIM;
                float xv   = x[idx];
                float nrm  = acc[i][j][r] + beta_f[m];
                out[idx]   = xv * rsqrtf(nrm);
            }
        }
    }
}

extern "C" void kernel_launch(void* const* d_in, const int* in_sizes, int n_in,
                              void* d_out, int out_size, void* d_ws, size_t ws_size,
                              hipStream_t stream) {
    const float* x       = (const float*)d_in[0];
    const float* beta_p  = (const float*)d_in[1];
    const float* gamma_p = (const float*)d_in[2];
    float* out = (float*)d_out;

    unsigned short* gamma_bf = (unsigned short*)d_ws;                       // 128 KB
    float*          beta_f   = (float*)((char*)d_ws + C_DIM * C_DIM * 2);   // 1 KB

    gdn_prep_kernel<<<(C_DIM * C_DIM + 255) / 256, 256, 0, stream>>>(
        beta_p, gamma_p, gamma_bf, beta_f);

    const size_t shbytes = (size_t)(C_DIM + TN) * LDA * 2;  // ~165 KB of 320 KB WGP LDS
    hipFuncSetAttribute((const void*)gdn_main_kernel,
                        hipFuncAttributeMaxDynamicSharedMemorySize, (int)shbytes);

    gdn_main_kernel<<<B_DIM * (L_DIM / TN), 256, shbytes, stream>>>(
        x, gamma_bf, beta_f, out);
}
